// MultiHeadAttention_16741782520180
// MI455X (gfx1250) — compile-verified, matrix operand order checked by reading, not run on hardware
//
#include <hip/hip_runtime.h>
#include <hip/hip_bf16.h>

// Problem constants (match reference): B=2, S=2048, D=1024, H=16, DK=64
#define Bc 2
#define Sc 2048
#define Dc 1024
#define Hc 16
#define DKc 64

typedef __attribute__((ext_vector_type(16))) __bf16        v16bf;
typedef __attribute__((ext_vector_type(8)))  float         v8f;
typedef __attribute__((ext_vector_type(4)))  unsigned int  u32x4;

union FragU { u32x4 u[2]; v16bf v; };
union PackU { u32x4 u; __bf16 e[8]; };

// Load a 16x32 bf16 WMMA operand fragment (A-layout; B uses the same layout
// with "row" = output column). ISA 16-bit A 16x32 layout: lane L holds row
// M=L%16; lanes 0-15 hold K in {0..7,16..23}, lanes 16-31 K in {8..15,24..31}.
__device__ __forceinline__ v16bf ldfrag(const __bf16* p, int ld) {
  int lane = threadIdx.x & 31;
  const __bf16* q = p + (size_t)(lane & 15) * ld + ((lane >> 4) << 3);
  FragU t;
  t.u[0] = *reinterpret_cast<const u32x4*>(q);       // K base+0..7
  t.u[1] = *reinterpret_cast<const u32x4*>(q + 16);  // K base+16..23
  return t.v;
}

__device__ __forceinline__ v8f wmma_bf16(v16bf a, v16bf b, v8f c) {
  return __builtin_amdgcn_wmma_f32_16x16x32_bf16(
      /*neg_a=*/false, a, /*neg_b=*/false, b,
      /*c_mod=*/(short)0, c, /*reuse_a=*/false, /*reuse_b=*/false);
}

// ---------------------------------------------------------------- conversions
__global__ void cvt_bf16_kernel(const float* __restrict__ in,
                                __bf16* __restrict__ out, int n) {
  int i = blockIdx.x * blockDim.x + threadIdx.x;
  if (i < n) out[i] = (__bf16)in[i];
}

// ---------------------------------------------------------------- GEMM
// Y = (X @ W^T + bias) * oscale.  X: [4096,1024] bf16 rm, W: [1024,1024] bf16
// rm (rows = output features -> feed B-fragments directly).
// MODE 0: out f32 flat [M, D]
// MODE 1: out bf16 head-split [B, H, S, DK]
// MODE 2: out bf16 head-split transposed [B, H, DK, S]   (for V)
// Block = 8 waves as 2(M)x4(N); wave tile 64x32 = 4x2 WMMA accumulators.
template <int MODE>
__global__ __launch_bounds__(256) void gemm_bf16_kernel(
    const __bf16* __restrict__ X, const __bf16* __restrict__ W,
    const float* __restrict__ bias, void* __restrict__ Y, float oscale) {
  const int K = Dc;
  int w    = threadIdx.x >> 5;
  int lane = threadIdx.x & 31;
  int m0 = blockIdx.y * 128 + (w >> 2) * 64;  // 2 waves along M
  int n0 = blockIdx.x * 128 + (w & 3) * 32;   // 4 waves along N

  v8f acc[4][2];
#pragma unroll
  for (int i = 0; i < 4; ++i)
#pragma unroll
    for (int jj = 0; jj < 2; ++jj) acc[i][jj] = (v8f){};

  for (int kk = 0; kk < K; kk += 32) {
    if (kk + 32 < K) {
      __builtin_prefetch(X + (size_t)m0 * K + kk + 32, 0, 0);
      __builtin_prefetch(W + (size_t)n0 * K + kk + 32, 0, 0);
    }
    v16bf a0 = ldfrag(X + (size_t)(m0 +  0) * K + kk, K);
    v16bf a1 = ldfrag(X + (size_t)(m0 + 16) * K + kk, K);
    v16bf a2 = ldfrag(X + (size_t)(m0 + 32) * K + kk, K);
    v16bf a3 = ldfrag(X + (size_t)(m0 + 48) * K + kk, K);
    v16bf b0 = ldfrag(W + (size_t)(n0 +  0) * K + kk, K);
    v16bf b1 = ldfrag(W + (size_t)(n0 + 16) * K + kk, K);
    acc[0][0] = wmma_bf16(a0, b0, acc[0][0]);
    acc[0][1] = wmma_bf16(a0, b1, acc[0][1]);
    acc[1][0] = wmma_bf16(a1, b0, acc[1][0]);
    acc[1][1] = wmma_bf16(a1, b1, acc[1][1]);
    acc[2][0] = wmma_bf16(a2, b0, acc[2][0]);
    acc[2][1] = wmma_bf16(a2, b1, acc[2][1]);
    acc[3][0] = wmma_bf16(a3, b0, acc[3][0]);
    acc[3][1] = wmma_bf16(a3, b1, acc[3][1]);
  }

  int colLane = lane & 15;
  int rowHi   = (lane >> 4) << 3;  // C-layout: VGPR r -> row r / r+8
#pragma unroll
  for (int ti = 0; ti < 4; ++ti) {
#pragma unroll
    for (int tj = 0; tj < 2; ++tj) {
      int n = n0 + tj * 16 + colLane;
      float bval = bias[n];
#pragma unroll
      for (int r = 0; r < 8; ++r) {
        int m = m0 + ti * 16 + rowHi + r;
        float val = (acc[ti][tj][r] + bval) * oscale;
        if (MODE == 0) {
          ((float*)Y)[(size_t)m * Dc + n] = val;
        } else {
          int h = n >> 6, dk = n & 63;
          int bb = m >> 11, s = m & (Sc - 1);
          if (MODE == 1)
            ((__bf16*)Y)[((size_t)(bb * Hc + h) * Sc + s) * DKc + dk] = (__bf16)val;
          else
            ((__bf16*)Y)[((size_t)(bb * Hc + h) * DKc + dk) * Sc + s] = (__bf16)val;
        }
      }
    }
  }
}

// ---------------------------------------------------------------- attention
// Transposed flash-attention. Each wave owns 32 query rows (two 16-query
// groups sharing K/V fragment loads) of one (b,h). We compute S^T = K Q^T so
// the WMMA C-layout puts queries across lanes and keys across VGPRs:
//   * row max / row sum over keys are in-lane + one shfl_xor(16)
//   * softmax stats m,l are one scalar per lane
//   * the C-layout of S^T equals the B-fragment layout of P^T elementwise,
//     so P feeds O^T = V^T P^T directly from registers (no LDS restage).
// Q is pre-scaled by 1/sqrt(DK) in its projection. Exactly one (diagonal)
// key tile needs the causal mask. Qh/Kh: [B,H,S,DK] bf16; Vt: [B,H,DK,S]
// bf16 (rows = dk -> A-fragments of V^T are contiguous); ctx: [B,S,D] bf16.
__global__ __launch_bounds__(128) void attn_kernel(
    const __bf16* __restrict__ Qh, const __bf16* __restrict__ Kh,
    const __bf16* __restrict__ Vt, __bf16* __restrict__ ctx) {
  int wv   = threadIdx.x >> 5;
  int lane = threadIdx.x & 31;
  int bh = blockIdx.x;
  int b = bh >> 4, h = bh & 15;
  int q0 = blockIdx.y * 128 + wv * 32;   // multiple of 32

  const __bf16* Qb = Qh + ((size_t)bh * Sc + q0) * DKc;
  const __bf16* Kb = Kh + (size_t)bh * Sc * DKc;
  const __bf16* Vb = Vt + (size_t)bh * DKc * Sc;

  int colLane = lane & 15;
  int hb      = (lane >> 4) << 3;  // which half of the 16 keys this lane holds
  int row0 = q0 + colLane;         // query row of group 0 for this lane
  int row1 = row0 + 16;            // query row of group 1

  // Q B-fragments (lane = query column) for the two query groups
  v16bf qa0 = ldfrag(Qb, DKc);
  v16bf qc0 = ldfrag(Qb + 32, DKc);
  v16bf qa1 = ldfrag(Qb + (size_t)16 * DKc, DKc);
  v16bf qc1 = ldfrag(Qb + (size_t)16 * DKc + 32, DKc);

  v8f o0[4] = {}, o1[4] = {};      // O^T accumulators: dk across VGPRs
  float m0 = -1e30f, l0 = 0.0f, m1 = -1e30f, l1 = 0.0f;

  // Online softmax over this lane's 16 keys of one score tile pair; returns
  // the P^T B-fragment for O^T += V^T P^T.
  auto softmax16 = [&](const v8f& sa, const v8f& sb, float& m, float& l,
                       v8f* o, int row, int key0, int key1,
                       bool masked) -> v16bf {
    float va[8], vb2[8];
#pragma unroll
    for (int r = 0; r < 8; ++r) { va[r] = sa[r]; vb2[r] = sb[r]; }
    if (masked) {
#pragma unroll
      for (int r = 0; r < 8; ++r) {
        if (key0 + r > row) va[r]  = -1e9f;
        if (key1 + r > row) vb2[r] = -1e9f;
      }
    }
    float mx = -1e30f;
#pragma unroll
    for (int r = 0; r < 8; ++r) mx = fmaxf(mx, fmaxf(va[r], vb2[r]));
    mx = fmaxf(mx, __shfl_xor(mx, 16, 32));   // merge complementary key half
    float mnew = fmaxf(m, mx);
    float corr = __expf(m - mnew);
    v16bf pf;
    float rs = 0.0f;
#pragma unroll
    for (int r = 0; r < 8; ++r) {
      float p0 = __expf(va[r] - mnew);
      float p1 = __expf(vb2[r] - mnew);
      rs += p0 + p1;
      pf[r]     = (__bf16)p0;   // B-frag elems 0..7  <-> key_local hb+0..7
      pf[8 + r] = (__bf16)p1;   // B-frag elems 8..15 <-> key_local hb+16..23
    }
    rs += __shfl_xor(rs, 16, 32);
    l = l * corr + rs;
    m = mnew;
#pragma unroll
    for (int t = 0; t < 4; ++t)
#pragma unroll
      for (int r = 0; r < 8; ++r) o[t][r] *= corr;
    return pf;
  };

  auto key_tile = [&](int j, bool masked) {
    // K A-fragments (rows = keys), shared by both query groups
    v16bf k0a = ldfrag(Kb + (size_t)j * DKc, DKc);
    v16bf k0b = ldfrag(Kb + (size_t)j * DKc + 32, DKc);
    v16bf k1a = ldfrag(Kb + (size_t)(j + 16) * DKc, DKc);
    v16bf k1b = ldfrag(Kb + (size_t)(j + 16) * DKc + 32, DKc);
    v8f s00 = {}, s01 = {}, s10 = {}, s11 = {};
    s00 = wmma_bf16(k0a, qa0, s00); s00 = wmma_bf16(k0b, qc0, s00);
    s01 = wmma_bf16(k1a, qa0, s01); s01 = wmma_bf16(k1b, qc0, s01);
    s10 = wmma_bf16(k0a, qa1, s10); s10 = wmma_bf16(k0b, qc1, s10);
    s11 = wmma_bf16(k1a, qa1, s11); s11 = wmma_bf16(k1b, qc1, s11);

    int key0 = j + hb, key1 = j + 16 + hb;
    v16bf pf0 = softmax16(s00, s01, m0, l0, o0, row0, key0, key1, masked);
    v16bf pf1 = softmax16(s10, s11, m1, l1, o1, row1, key0, key1, masked);

    // V^T A-fragments (rows = dk), shared by both query groups
#pragma unroll
    for (int t = 0; t < 4; ++t) {
      v16bf vf = ldfrag(Vb + (size_t)(t * 16) * Sc + j, Sc);
      o0[t] = wmma_bf16(vf, pf0, o0[t]);
      o1[t] = wmma_bf16(vf, pf1, o1[t]);
    }
  };

  // Tiles fully below the diagonal need no mask; exactly one diagonal tile.
  for (int j = 0; j < q0; j += 32) key_tile(j, false);
  key_tile(q0, true);

  // epilogue: ctx[b, s, h*DK + dk] = O / l  -- 8 consecutive dk per b128 store
  float inv0 = 1.0f / l0, inv1 = 1.0f / l1;
  __bf16* cp0 = ctx + ((size_t)b * Sc + row0) * Dc + h * DKc;
  __bf16* cp1 = ctx + ((size_t)b * Sc + row1) * Dc + h * DKc;
#pragma unroll
  for (int t = 0; t < 4; ++t) {
    PackU pk;
#pragma unroll
    for (int r = 0; r < 8; ++r) pk.e[r] = (__bf16)(o0[t][r] * inv0);
    *reinterpret_cast<u32x4*>(cp0 + t * 16 + hb) = pk.u;
#pragma unroll
    for (int r = 0; r < 8; ++r) pk.e[r] = (__bf16)(o1[t][r] * inv1);
    *reinterpret_cast<u32x4*>(cp1 + t * 16 + hb) = pk.u;
  }
}

// ---------------------------------------------------------------- launch
extern "C" void kernel_launch(void* const* d_in, const int* in_sizes, int n_in,
                              void* d_out, int out_size, void* d_ws, size_t ws_size,
                              hipStream_t stream) {
  (void)in_sizes; (void)n_in; (void)out_size; (void)ws_size;
  const float* q  = (const float*)d_in[0];
  const float* k  = (const float*)d_in[1];
  const float* v  = (const float*)d_in[2];
  // d_in[3] = causal mask (tril) -- computed analytically in the kernel
  const float* Wq = (const float*)d_in[4];
  const float* bq = (const float*)d_in[5];
  const float* Wk = (const float*)d_in[6];
  const float* bk = (const float*)d_in[7];
  const float* Wv = (const float*)d_in[8];
  const float* bv = (const float*)d_in[9];
  const float* Wo = (const float*)d_in[10];
  const float* bo = (const float*)d_in[11];
  float* out = (float*)d_out;

  const size_t MN = (size_t)Bc * Sc * Dc;  // 4M elements
  const size_t WN = (size_t)Dc * Dc;       // 1M elements
  __bf16* p = (__bf16*)d_ws;
  __bf16* qbuf = p; p += MN;
  __bf16* kbuf = p; p += MN;
  __bf16* vbuf = p; p += MN;
  __bf16* wqb  = p; p += WN;
  __bf16* wkb  = p; p += WN;
  __bf16* wvb  = p; p += WN;
  __bf16* wob  = p; p += WN;
  __bf16* Qh   = p; p += MN;  // [B,H,S,DK]  (pre-scaled by 1/sqrt(DK))
  __bf16* Kh   = p; p += MN;  // [B,H,S,DK]
  __bf16* Vt   = p; p += MN;  // [B,H,DK,S]
  __bf16* ctx  = p; p += MN;  // [B,S,D]

  const int nM = (int)MN, nW = (int)WN;
  cvt_bf16_kernel<<<(nM + 255) / 256, 256, 0, stream>>>(q, qbuf, nM);
  cvt_bf16_kernel<<<(nM + 255) / 256, 256, 0, stream>>>(k, kbuf, nM);
  cvt_bf16_kernel<<<(nM + 255) / 256, 256, 0, stream>>>(v, vbuf, nM);
  cvt_bf16_kernel<<<(nW + 255) / 256, 256, 0, stream>>>(Wq, wqb, nW);
  cvt_bf16_kernel<<<(nW + 255) / 256, 256, 0, stream>>>(Wk, wkb, nW);
  cvt_bf16_kernel<<<(nW + 255) / 256, 256, 0, stream>>>(Wv, wvb, nW);
  cvt_bf16_kernel<<<(nW + 255) / 256, 256, 0, stream>>>(Wo, wob, nW);

  dim3 gg(Dc / 128, (Bc * Sc) / 128);  // (8, 32)
  gemm_bf16_kernel<1><<<gg, 256, 0, stream>>>(qbuf, wqb, bq, Qh, 0.125f);
  gemm_bf16_kernel<1><<<gg, 256, 0, stream>>>(kbuf, wkb, bk, Kh, 1.0f);
  gemm_bf16_kernel<2><<<gg, 256, 0, stream>>>(vbuf, wvb, bv, Vt, 1.0f);

  dim3 ga(Bc * Hc, Sc / 128);  // (32, 16)
  attn_kernel<<<ga, 128, 0, stream>>>(Qh, Kh, Vt, ctx);

  gemm_bf16_kernel<0><<<gg, 256, 0, stream>>>(ctx, wob, bo, (void*)out, 1.0f);
}
